// DecoderRNN_23450521436221
// MI455X (gfx1250) — compile-verified, matrix operand order checked by reading, not run on hardware
//
#include <hip/hip_runtime.h>

// ---------------- sizes ----------------
#define VOCAB 50257
#define EMB   512
#define HID   512
#define NLAY  2
#define BB    32
#define TT    32
#define SS    128
#define SOS_ID 1

typedef __attribute__((ext_vector_type(16))) __bf16 v16bf;
typedef __attribute__((ext_vector_type(8)))  float  v8f;

union FragU { unsigned int u[8]; v16bf v; };

// LDS row stride in uints: 80B rows -> b128-aligned, banks spread (20 mod 64)
#define LPITCH 20

static __device__ __forceinline__ unsigned short f2bf(float f) {
    unsigned int u = __float_as_uint(f);
    unsigned int r = u + 0x7FFFu + ((u >> 16) & 1u);   // round-to-nearest-even
    return (unsigned short)(r >> 16);
}

// -------- generic elementwise f32 -> bf16 --------
__global__ void cvt_bf16_kernel(const float* __restrict__ in,
                                unsigned short* __restrict__ out, int n) {
    int i = blockIdx.x * 256 + threadIdx.x;
    if (i < n) out[i] = f2bf(in[i]);
}

// -------- encoder_outputs (S,B,H) -> enc_bf16 (B,S,H) --------
__global__ void enc_transpose_kernel(const float* __restrict__ enc,
                                     unsigned short* __restrict__ out) {
    int i = blockIdx.x * 256 + threadIdx.x;       // B*S*H
    if (i >= BB * SS * HID) return;
    int h = i & (HID - 1);
    int s = (i >> 9) & (SS - 1);
    int b = i >> 16;
    out[i] = f2bf(enc[((size_t)s * BB + b) * HID + h]);
}

// -------- h init: decoder_hidden -> h_f32 + h_bf16 --------
__global__ void init_h_kernel(const float* __restrict__ dh,
                              float* __restrict__ h,
                              unsigned short* __restrict__ hbf) {
    int i = blockIdx.x * 256 + threadIdx.x;       // L*B*H
    if (i >= NLAY * BB * HID) return;
    float v = dh[i];
    h[i] = v;
    hbf[i] = f2bf(v);
}

// -------- embedding gather (teacher forcing, bf16 out) --------
__global__ void embed_kernel(const int* __restrict__ tgt,
                             const float* __restrict__ embW,
                             unsigned short* __restrict__ xbf, int t) {
    int i = blockIdx.x * 256 + threadIdx.x;       // B*E
    if (i >= BB * EMB) return;
    int b = i >> 9, e = i & (EMB - 1);
    int tok = (t == 0) ? SOS_ID : tgt[b * TT + (t - 1)];
    xbf[i] = f2bf(embW[(size_t)tok * EMB + e]);
}

// fragment K gather: vgpr v holds K pair at uint index (v>=4?8:0) + half*4 + (v&3)
#define KU(v, half) (((v) >= 4 ? 8 : 0) + (half) * 4 + ((v) & 3))

// ======== small-M WMMA GEMM: D[M,N] = A[M,K]*B[N,K]^T + bias ========
// M multiple of 32, N multiple of 128, K multiple of 32.
// grid = (N/128, M/32), block = 256. Double-buffered LDS, 1 barrier/chunk.
__global__ __launch_bounds__(256) void gemm_bf16_wmma_m32(
    const unsigned short* __restrict__ A,   // M x K (bf16)
    const unsigned short* __restrict__ Bm,  // N x K (bf16)
    const float* __restrict__ bias,         // N
    float* __restrict__ D,                  // M x N
    int M, int N, int K)
{
    __shared__ unsigned int As[2][32][LPITCH];
    __shared__ unsigned int Bs[2][128][LPITCH];

    const int tid  = threadIdx.x;
    const int lane = tid & 31;
    const int wv   = tid >> 5;              // 0..7 -> one 16-col tile per wave
    const int half = lane >> 4;
    const int lr   = lane & 15;
    const int m0   = blockIdx.y * 32;
    const int n0   = blockIdx.x * 128;

    // staging: A tile = 128 quads (tid<128: 1 each); B tile = 512 quads (2 each)
    const int ar = tid >> 2, ac = tid & 3;               // A quad coords
    uint4 ra, rb[2];

    auto gload = [&](int k0) {
        if (tid < 128)
            ra = *(const uint4*)(A + (size_t)(m0 + ar) * K + k0 + ac * 8);
#pragma unroll
        for (int j = 0; j < 2; ++j) {
            int q = tid + j * 256;
            int r = q >> 2, c = q & 3;
            rb[j] = *(const uint4*)(Bm + (size_t)(n0 + r) * K + k0 + c * 8);
        }
    };
    auto stash = [&](int bi) {
        if (tid < 128)
            *(uint4*)&As[bi][ar][ac * 4] = ra;
#pragma unroll
        for (int j = 0; j < 2; ++j) {
            int q = tid + j * 256;
            int r = q >> 2, c = q & 3;
            *(uint4*)&Bs[bi][r][c * 4] = rb[j];
        }
    };

    v8f acc0 = {}; v8f acc1 = {};

    const int nK = K >> 5;
    gload(0);
    stash(0);
    __syncthreads();

    for (int kc = 0; kc < nK; ++kc) {
        const int cur = kc & 1;
        if (kc + 1 < nK) gload((kc + 1) << 5);   // overlap with compute below

        FragU fa0, fa1, fb;
#pragma unroll
        for (int v = 0; v < 8; ++v) {
            int ku = KU(v, half);
            fa0.u[v] = As[cur][lr][ku];
            fa1.u[v] = As[cur][16 + lr][ku];
            fb.u[v]  = Bs[cur][wv * 16 + lr][ku];
        }
        acc0 = __builtin_amdgcn_wmma_f32_16x16x32_bf16(false, fa0.v, false, fb.v,
                                                       (short)0, acc0, false, false);
        acc1 = __builtin_amdgcn_wmma_f32_16x16x32_bf16(false, fa1.v, false, fb.v,
                                                       (short)0, acc1, false, false);
        if (kc + 1 < nK) stash(1 - cur);
        __syncthreads();
    }

    int col = n0 + wv * 16 + lr;
    float bv = bias[col];
#pragma unroll
    for (int i = 0; i < 8; ++i) {
        int mr = half * 8 + i;
        D[(size_t)(m0 + mr) * N + col]      = acc0[i] + bv;
        D[(size_t)(m0 + 16 + mr) * N + col] = acc1[i] + bv;
    }
}

// ======== big-M WMMA GEMM: D[M,N] = A[M,K]*B[N,K]^T + bias ========
// M multiple of 128; N arbitrary (clamped, branch-free); K multiple of 32.
// grid = (ceil(N/128), M/128), block = 256 (8 waves, 4x2 layout).
// Each wave: 2 m-tiles x 4 n-tiles -> 8 v_wmma per K-chunk.
// Double-buffered LDS ping-pong, 1 barrier per chunk, b128 staging.
__global__ __launch_bounds__(256) void gemm_bf16_wmma_m128(
    const unsigned short* __restrict__ A,   // M x K (bf16)
    const unsigned short* __restrict__ Bm,  // N x K (bf16)
    const float* __restrict__ bias,         // N
    float* __restrict__ D,                  // M x N
    int M, int N, int K)
{
    __shared__ unsigned int As[2][128][LPITCH];
    __shared__ unsigned int Bs[2][128][LPITCH];

    const int tid  = threadIdx.x;
    const int lane = tid & 31;
    const int wv   = tid >> 5;
    const int half = lane >> 4;
    const int lr   = lane & 15;
    const int m0   = blockIdx.y * 128;
    const int n0   = blockIdx.x * 128;
    const int wm   = (wv & 3) * 32;         // wave row offset (2 m-tiles)
    const int wn   = (wv >> 2) * 64;        // wave col offset (4 n-tiles)

    // staging: each tile = 512 quads -> 2 quads/thread per tile
    uint4 ra[2], rb[2];

    auto gload = [&](int k0) {
#pragma unroll
        for (int j = 0; j < 2; ++j) {
            int q = tid + j * 256;
            int r = q >> 2, c = q & 3;
            ra[j] = *(const uint4*)(A + (size_t)(m0 + r) * K + k0 + c * 8);
            int n = n0 + r;
            int nc = n < N ? n : (N - 1);
            uint4 v = *(const uint4*)(Bm + (size_t)nc * K + k0 + c * 8);
            if (n >= N) v = make_uint4(0u, 0u, 0u, 0u);   // cndmask, no exec branch
            rb[j] = v;
        }
    };
    auto stash = [&](int bi) {
#pragma unroll
        for (int j = 0; j < 2; ++j) {
            int q = tid + j * 256;
            int r = q >> 2, c = q & 3;
            *(uint4*)&As[bi][r][c * 4] = ra[j];
            *(uint4*)&Bs[bi][r][c * 4] = rb[j];
        }
    };

    v8f acc[2][4] = {};

    const int nK = K >> 5;
    gload(0);
    stash(0);
    __syncthreads();

    for (int kc = 0; kc < nK; ++kc) {
        const int cur = kc & 1;
        if (kc + 1 < nK) gload((kc + 1) << 5);   // global b128 loads in flight

        // prefetch chunk kc+2 of the B stream (global_prefetch_b8)
        if (kc + 2 < nK) {
            int prow = n0 + (tid >> 1);
            int prc  = prow < N ? prow : (N - 1);
            __builtin_prefetch(Bm + (size_t)prc * K + ((kc + 2) << 5) + (tid & 1) * 16, 0, 1);
        }

        FragU fa[2], fb[4];
#pragma unroll
        for (int v = 0; v < 8; ++v) {
            int ku = KU(v, half);
#pragma unroll
            for (int i = 0; i < 2; ++i) fa[i].u[v] = As[cur][wm + i * 16 + lr][ku];
#pragma unroll
            for (int j = 0; j < 4; ++j) fb[j].u[v] = Bs[cur][wn + j * 16 + lr][ku];
        }
#pragma unroll
        for (int i = 0; i < 2; ++i)
#pragma unroll
            for (int j = 0; j < 4; ++j)
                acc[i][j] = __builtin_amdgcn_wmma_f32_16x16x32_bf16(
                    false, fa[i].v, false, fb[j].v, (short)0, acc[i][j], false, false);

        if (kc + 1 < nK) stash(1 - cur);
        __syncthreads();
    }

    // epilogue: D row = m0 + wm + i*16 + half*8 + e ; col = n0 + wn + j*16 + lr
#pragma unroll
    for (int j = 0; j < 4; ++j) {
        int col = n0 + wn + j * 16 + lr;
        if (col < N) {
            float bv = bias[col];
#pragma unroll
            for (int i = 0; i < 2; ++i) {
#pragma unroll
                for (int e = 0; e < 8; ++e) {
                    int row = m0 + wm + i * 16 + half * 8 + e;
                    D[(size_t)row * N + col] = acc[i][j][e] + bv;
                }
            }
        }
    }
}

// -------- GRU gate pointwise: r,z,n + state update --------
__global__ void gru_pointwise(const float* __restrict__ gx,
                              const float* __restrict__ gh,
                              float* __restrict__ h,             // 32x512 (this layer)
                              unsigned short* __restrict__ hbf,  // 32x512 bf16
                              float* __restrict__ xattn,         // null unless last layer
                              unsigned short* __restrict__ catbf)// null unless last layer
{
    int i = blockIdx.x * 256 + threadIdx.x;   // B*H
    if (i >= BB * HID) return;
    int b = i >> 9, hh = i & (HID - 1);
    size_t g0 = (size_t)b * 3 * HID + hh;
    float r = 1.0f / (1.0f + __expf(-(gx[g0] + gh[g0])));
    float z = 1.0f / (1.0f + __expf(-(gx[g0 + HID] + gh[g0 + HID])));
    float n = tanhf(gx[g0 + 2 * HID] + r * gh[g0 + 2 * HID]);
    float hn = (1.0f - z) * n + z * h[i];
    h[i] = hn;
    unsigned short hb = f2bf(hn);
    hbf[i] = hb;
    if (xattn) {
        xattn[i] = hn;
        catbf[(size_t)b * (2 * HID) + HID + hh] = hb;   // x half of [ctx, x]
    }
}

// -------- Luong attention: scores -> softmax -> ctx (one block per batch) --------
__global__ __launch_bounds__(128) void attn_kernel(
    const float* __restrict__ xattn,       // B x H
    const float* __restrict__ enc_proj,    // B x S x H
    const float* __restrict__ enc_src,     // S x B x H (original)
    const unsigned char* __restrict__ mask,// B x S
    unsigned short* __restrict__ catbf)    // B x 2H (writes ctx half)
{
    __shared__ float xs[HID];
    __shared__ float sc[SS];
    __shared__ float red[SS];
    int b = blockIdx.x, tid = threadIdx.x;

    for (int i = tid; i < HID; i += 128) xs[i] = xattn[(size_t)b * HID + i];
    __syncthreads();

    {
        const float* ep = enc_proj + ((size_t)b * SS + tid) * HID;
        float s = 0.f;
        for (int hh = 0; hh < HID; ++hh) s += xs[hh] * ep[hh];
        if (mask[b * SS + tid]) s = -1e9f;
        sc[tid] = s;
    }
    __syncthreads();

    red[tid] = sc[tid]; __syncthreads();
    for (int off = 64; off > 0; off >>= 1) {
        if (tid < off) red[tid] = fmaxf(red[tid], red[tid + off]);
        __syncthreads();
    }
    float mx = red[0]; __syncthreads();
    float e = __expf(sc[tid] - mx);
    red[tid] = e; __syncthreads();
    for (int off = 64; off > 0; off >>= 1) {
        if (tid < off) red[tid] += red[tid + off];
        __syncthreads();
    }
    float inv = 1.0f / red[0]; __syncthreads();
    sc[tid] = e * inv;
    __syncthreads();

    for (int j = 0; j < 4; ++j) {
        int hh = tid + j * 128;
        float c = 0.f;
        for (int s = 0; s < SS; ++s)
            c += sc[s] * enc_src[((size_t)s * BB + b) * HID + hh];
        catbf[(size_t)b * (2 * HID) + hh] = f2bf(c);
    }
}

// -------- tanh epilogue of Wc GEMM -> concats_bf16 row t --------
__global__ void tanh_store_kernel(const float* __restrict__ tmp,
                                  unsigned short* __restrict__ concats, int t) {
    int i = blockIdx.x * 256 + threadIdx.x;   // B*H
    if (i >= BB * HID) return;
    concats[(size_t)t * BB * HID + i] = f2bf(tanhf(tmp[i]));
}

// -------- final h copy into d_out tail --------
__global__ void copy_h_kernel(const float* __restrict__ h, float* __restrict__ dst) {
    int i = blockIdx.x * 256 + threadIdx.x;   // L*B*H
    if (i < NLAY * BB * HID) dst[i] = h[i];
}

extern "C" void kernel_launch(void* const* d_in, const int* in_sizes, int n_in,
                              void* d_out, int out_size, void* d_ws, size_t ws_size,
                              hipStream_t stream) {
    const int*   tgt  = (const int*)d_in[0];
    const float* dh   = (const float*)d_in[1];
    const float* enc  = (const float*)d_in[2];
    const unsigned char* mask = (const unsigned char*)d_in[3];
    const float* embW = (const float*)d_in[4];
    const float* Wih  = (const float*)d_in[5];
    const float* Whh  = (const float*)d_in[6];
    const float* bih  = (const float*)d_in[7];
    const float* bhh  = (const float*)d_in[8];
    const float* Wa   = (const float*)d_in[9];
    const float* ba   = (const float*)d_in[10];
    const float* Wc   = (const float*)d_in[11];
    const float* bc   = (const float*)d_in[12];
    const float* Ws   = (const float*)d_in[13];
    const float* bs   = (const float*)d_in[14];
    float* out = (float*)d_out;

    // ---- workspace carve-up (256B aligned) ----
    char* p = (char*)d_ws;
    auto take = [&](size_t bytes) -> char* {
        char* r = p; p += (bytes + 255) & ~(size_t)255; return r;
    };
    unsigned short* Ws_bf   = (unsigned short*)take((size_t)VOCAB * HID * 2);
    unsigned short* Wih_bf  = (unsigned short*)take((size_t)NLAY * 3 * HID * EMB * 2);
    unsigned short* Whh_bf  = (unsigned short*)take((size_t)NLAY * 3 * HID * HID * 2);
    unsigned short* Wc_bf   = (unsigned short*)take((size_t)HID * 2 * HID * 2);
    unsigned short* Wa_bf   = (unsigned short*)take((size_t)HID * HID * 2);
    unsigned short* enc_bf  = (unsigned short*)take((size_t)BB * SS * HID * 2);
    float*          encproj = (float*)take((size_t)BB * SS * HID * 4);
    unsigned short* conc_bf = (unsigned short*)take((size_t)TT * BB * HID * 2);
    unsigned short* x_bf    = (unsigned short*)take((size_t)BB * EMB * 2);
    float*          h_f     = (float*)take((size_t)NLAY * BB * HID * 4);
    unsigned short* h_bf    = (unsigned short*)take((size_t)NLAY * BB * HID * 2);
    float*          gx      = (float*)take((size_t)BB * 3 * HID * 4);
    float*          gh      = (float*)take((size_t)BB * 3 * HID * 4);
    float*          xattn   = (float*)take((size_t)BB * HID * 4);
    unsigned short* cat_bf  = (unsigned short*)take((size_t)BB * 2 * HID * 2);
    float*          tmpcat  = (float*)take((size_t)BB * HID * 4);

    auto blocks = [](int n) { return (n + 255) / 256; };

    // ---- one-time conversions (re-run every call; deterministic) ----
    cvt_bf16_kernel<<<blocks(VOCAB * HID), 256, 0, stream>>>(Ws, Ws_bf, VOCAB * HID);
    cvt_bf16_kernel<<<blocks(NLAY * 3 * HID * EMB), 256, 0, stream>>>(Wih, Wih_bf, NLAY * 3 * HID * EMB);
    cvt_bf16_kernel<<<blocks(NLAY * 3 * HID * HID), 256, 0, stream>>>(Whh, Whh_bf, NLAY * 3 * HID * HID);
    cvt_bf16_kernel<<<blocks(HID * 2 * HID), 256, 0, stream>>>(Wc, Wc_bf, HID * 2 * HID);
    cvt_bf16_kernel<<<blocks(HID * HID), 256, 0, stream>>>(Wa, Wa_bf, HID * HID);
    enc_transpose_kernel<<<blocks(BB * SS * HID), 256, 0, stream>>>(enc, enc_bf);
    init_h_kernel<<<blocks(NLAY * BB * HID), 256, 0, stream>>>(dh, h_f, h_bf);

    // ---- enc_proj = enc @ Wa^T + ba : (B*S=4096, 512) x (512, 512) ----
    {
        dim3 g((HID + 127) / 128, (BB * SS) / 128);
        gemm_bf16_wmma_m128<<<g, 256, 0, stream>>>(enc_bf, Wa_bf, ba, encproj,
                                                   BB * SS, HID, HID);
    }

    // ---- sequential decode, stream-ordered kernels per step ----
    for (int t = 0; t < TT; ++t) {
        embed_kernel<<<blocks(BB * EMB), 256, 0, stream>>>(tgt, embW, x_bf, t);
        for (int l = 0; l < NLAY; ++l) {
            const unsigned short* in_bf = (l == 0) ? x_bf : (h_bf + (size_t)(l - 1) * BB * HID);
            dim3 g((3 * HID) / 128, 1);
            gemm_bf16_wmma_m32<<<g, 256, 0, stream>>>(
                in_bf, Wih_bf + (size_t)l * 3 * HID * EMB, bih + (size_t)l * 3 * HID,
                gx, BB, 3 * HID, EMB);
            gemm_bf16_wmma_m32<<<g, 256, 0, stream>>>(
                h_bf + (size_t)l * BB * HID, Whh_bf + (size_t)l * 3 * HID * HID,
                bhh + (size_t)l * 3 * HID, gh, BB, 3 * HID, HID);
            bool last = (l == NLAY - 1);
            gru_pointwise<<<blocks(BB * HID), 256, 0, stream>>>(
                gx, gh, h_f + (size_t)l * BB * HID, h_bf + (size_t)l * BB * HID,
                last ? xattn : nullptr, last ? cat_bf : nullptr);
        }
        attn_kernel<<<BB, 128, 0, stream>>>(xattn, encproj, enc, mask, cat_bf);
        {
            dim3 g(HID / 128, 1);
            gemm_bf16_wmma_m32<<<g, 256, 0, stream>>>(cat_bf, Wc_bf, bc, tmpcat,
                                                      BB, HID, 2 * HID);
        }
        tanh_store_kernel<<<blocks(BB * HID), 256, 0, stream>>>(tmpcat, conc_bf, t);
    }

    // ---- dominant GEMM: logits (T*B=1024, V=50257) = concats @ Ws^T + bs ----
    {
        dim3 g((VOCAB + 127) / 128, (TT * BB) / 128);
        gemm_bf16_wmma_m128<<<g, 256, 0, stream>>>(conc_bf, Ws_bf, bs, out,
                                                   TT * BB, VOCAB, HID);
    }
    // ---- h_final tail ----
    copy_h_kernel<<<blocks(NLAY * BB * HID), 256, 0, stream>>>(
        h_f, out + (size_t)TT * BB * VOCAB);
}
